// MultiHeadAttention_49134425866549
// MI455X (gfx1250) — matrix operand order repaired and checked by reading, not compiled, not run on hardware
//
#include <hip/hip_runtime.h>
#include <hip/hip_bf16.h>
#include <stdint.h>

// Problem constants (from reference): B=2, T=2048, E=1024, H=16, D=64
#define B_  2
#define T_  2048
#define E_  1024
#define H_  16
#define D_  64
#define N3_ 3072
#define M_  (B_*T_)   // 4096 rows for the GEMMs

#define AS1 __attribute__((address_space(1)))
#define AS3 __attribute__((address_space(3)))

typedef __attribute__((ext_vector_type(16))) __bf16 v16bf;
typedef __attribute__((ext_vector_type(8)))  float  v8f;
typedef __attribute__((ext_vector_type(4)))  int    v4i_t;

union V16 {
  v16bf bf;
  unsigned short u[16];
};

__device__ __forceinline__ unsigned short f32_to_bf16(float f) {
  unsigned int u = __float_as_uint(f);
  u += 0x7FFFu + ((u >> 16) & 1u);     // round-to-nearest-even
  return (unsigned short)(u >> 16);
}

__device__ __forceinline__ v8f vzero8() {
  v8f r;
  #pragma unroll
  for (int i = 0; i < 8; ++i) r[i] = 0.0f;
  return r;
}

// Async global -> LDS 16-byte copy (per active lane), tracked by ASYNCcnt.
// Probed prototype: (int4 __device__* gsrc, int4 __shared__* ldst, imm, imm)
__device__ __forceinline__ void async_copy_b128(const void* gsrc, void* ldst) {
#if __has_builtin(__builtin_amdgcn_global_load_async_to_lds_b128)
  __builtin_amdgcn_global_load_async_to_lds_b128(
      (AS1 v4i_t*)gsrc, (AS3 v4i_t*)ldst, 0, 0);
#else
  const AS1 void* g = (const AS1 void*)gsrc;
  AS3 void*       l = (AS3 void*)ldst;
  asm volatile("global_load_async_to_lds_b128 %0, %1, off"
               :: "v"(l), "v"(g) : "memory");
#endif
}

__device__ __forceinline__ void wait_async0() {
  asm volatile("s_wait_asynccnt 0" ::: "memory");
}

// ---------------------------------------------------------------- f32 -> bf16
__global__ void f2bf_kernel(const float* __restrict__ src,
                            unsigned short* __restrict__ dst, int n) {
  int i = blockIdx.x * 256 + threadIdx.x;
  if (i < n) dst[i] = f32_to_bf16(src[i]);
}

// ------------------------------------------------------------- bf16 GEMM + bias
// C[M_, N] = A[M_,1024] * W[1024,N] + bias.  Workgroup = 64x64 tile, 8 waves,
// wave (wm,wn) computes a 16x32 sub-tile as two v_wmma_f32_16x16x32_bf16 per
// 32-wide K step.  mode 0: scatter bf16 into Q/K/V [B,H,T,D].  mode 1: fp32 out.
__global__ __launch_bounds__(256) void gemm_bf16_kernel(
    const unsigned short* __restrict__ A,
    const unsigned short* __restrict__ W,
    const float* __restrict__ bias,
    int N, int mode,
    unsigned short* __restrict__ Qh,
    unsigned short* __restrict__ Kh,
    unsigned short* __restrict__ Vh,
    float* __restrict__ outF) {
  __shared__ __attribute__((aligned(16))) unsigned short As[64][32];
  __shared__ __attribute__((aligned(16))) unsigned short BsT[64][32];

  const int tid  = threadIdx.x;
  const int wave = tid >> 5, lane = tid & 31;
  const int wm = wave & 3, wn = wave >> 2;
  const int half = lane >> 4, l16 = lane & 15;
  const int bM = blockIdx.x * 64, bN = blockIdx.y * 64;

  v8f acc0 = vzero8(), acc1 = vzero8();

  for (int k0 = 0; k0 < 1024; k0 += 32) {
    __syncthreads();
    // A tile 64x32 bf16 (row-major): async copy, 16 bytes per thread (4 KB total)
    {
      int row = tid >> 2;            // 64 rows
      int seg = tid & 3;             // 4 x 16B per 64B row
      async_copy_b128(A + (size_t)(bM + row) * 1024 + k0 + seg * 8,
                      ((char*)As) + tid * 16);
    }
    // W tile 32x64, stored N-major (transposed) so B-operand K-pairs are contiguous
    #pragma unroll
    for (int j = 0; j < 8; ++j) {
      int idx = tid * 8 + j;                // 2048 ushorts
      int kk = idx >> 6, n = idx & 63;
      BsT[n][kk] = W[(size_t)(k0 + kk) * N + bN + n];
    }
    wait_async0();
    __syncthreads();

    // A operand: lane = row M, VGPR i -> K = half*16 + 2i
    V16 a, b0, b1;
    #pragma unroll
    for (int i = 0; i < 8; ++i) {
      int kk = (half << 4) + (i << 1);
      a.u[2*i]   = As[wm*16 + l16][kk];
      a.u[2*i+1] = As[wm*16 + l16][kk+1];
    }
    // B operand: lane = col N (half selects K 0-15 / 16-31), VGPR i -> K = half*16+2i
    #pragma unroll
    for (int i = 0; i < 8; ++i) {
      int kk = (half << 4) + (i << 1);
      b0.u[2*i]   = BsT[wn*32 + l16][kk];
      b0.u[2*i+1] = BsT[wn*32 + l16][kk+1];
      b1.u[2*i]   = BsT[wn*32 + 16 + l16][kk];
      b1.u[2*i+1] = BsT[wn*32 + 16 + l16][kk+1];
    }
    acc0 = __builtin_amdgcn_wmma_f32_16x16x32_bf16(false, a.bf, false, b0.bf,
                                                   (short)0, acc0, false, false);
    acc1 = __builtin_amdgcn_wmma_f32_16x16x32_bf16(false, a.bf, false, b1.bf,
                                                   (short)0, acc1, false, false);
  }

  // Epilogue. C layout: VGPR i -> row m = i + 8*half, col n = lane%16
  #pragma unroll
  for (int i = 0; i < 8; ++i) {
    int r  = bM + wm * 16 + i + 8 * half;
    int c0 = bN + wn * 32 + l16;
    int c1 = c0 + 16;
    float v0 = acc0[i] + bias[c0];
    float v1 = acc1[i] + bias[c1];
    if (mode == 0) {
      int bb = r >> 11, t = r & (T_ - 1);
      {
        int which = c0 >> 10, e = c0 & 1023, hh = e >> 6, d = e & 63;
        unsigned short* dst = which == 0 ? Qh : (which == 1 ? Kh : Vh);
        dst[(((size_t)bb * H_ + hh) * T_ + t) * D_ + d] = f32_to_bf16(v0);
      }
      {
        int which = c1 >> 10, e = c1 & 1023, hh = e >> 6, d = e & 63;
        unsigned short* dst = which == 0 ? Qh : (which == 1 ? Kh : Vh);
        dst[(((size_t)bb * H_ + hh) * T_ + t) * D_ + d] = f32_to_bf16(v1);
      }
    } else {
      outF[(size_t)r * N + c0] = v0;
      outF[(size_t)r * N + c1] = v1;
    }
  }
}

// ------------------------------------------------------------------ attention
// One workgroup per (b,h, 128 q rows); 8 waves, each owning 16 q rows.
// Pass 1: online softmax stats (row max / row sum) over streamed K tiles.
// Pass 2: recompute scores, write normalized probs to attn, accumulate P*V.
// K tiles are streamed with async global->LDS copies (ASYNCcnt).
__global__ __launch_bounds__(256) void attn_kernel(
    const unsigned short* __restrict__ Qh,
    const unsigned short* __restrict__ Kh,
    const unsigned short* __restrict__ Vh,
    const unsigned char* __restrict__ maskb,
    float* __restrict__ attn,
    unsigned short* __restrict__ Octx) {
  __shared__ __attribute__((aligned(16))) unsigned short Ks[32][64];   // [key][d]
  __shared__ __attribute__((aligned(16))) unsigned short VsT[64][32];  // [d][key]
  __shared__ __attribute__((aligned(16))) unsigned short Ps[8][16][32];// per-wave P

  const int tid  = threadIdx.x;
  const int wave = tid >> 5, lane = tid & 31;
  const int half = lane >> 4, l16 = lane & 15;
  const int qblk = blockIdx.x, bh = blockIdx.y;
  const int b = bh >> 4, h = bh & 15;
  const int q0 = qblk * 128 + wave * 16;

  const unsigned short* Qb = Qh + ((size_t)bh * T_ + q0) * D_;
  const unsigned short* Kb = Kh + (size_t)bh * T_ * D_;
  const unsigned short* Vb = Vh + (size_t)bh * T_ * D_;
  const unsigned char* mrow = maskb + (size_t)b * T_;

  // Q as two pre-gathered A operands (d chunks 0-31, 32-63), kept in registers.
  V16 qa[2];
  #pragma unroll
  for (int c = 0; c < 2; ++c)
    #pragma unroll
    for (int i = 0; i < 8; ++i) {
      int d = c * 32 + (half << 4) + (i << 1);
      const unsigned short* p = Qb + (size_t)l16 * D_ + d;
      qa[c].u[2*i] = p[0]; qa[c].u[2*i+1] = p[1];
    }

  float rmax[8], rsum[8];
  #pragma unroll
  for (int i = 0; i < 8; ++i) { rmax[i] = -3.0e38f; rsum[i] = 0.0f; }

  // ---------------- pass 1: softmax statistics ----------------
  for (int kt = 0; kt < T_; kt += 32) {
    __syncthreads();
    // K tile 32x64 bf16 = 4 KB: one async b128 per thread
    async_copy_b128(Kb + (size_t)kt * D_ + tid * 8, ((char*)Ks) + tid * 16);
    wait_async0();
    __syncthreads();

    v8f s0 = vzero8(), s1 = vzero8();
    #pragma unroll
    for (int c = 0; c < 2; ++c) {
      V16 kb0, kb1;
      #pragma unroll
      for (int i = 0; i < 8; ++i) {
        int d = c * 32 + (half << 4) + (i << 1);
        const unsigned short* p0 = &Ks[l16][d];
        const unsigned short* p1 = &Ks[16 + l16][d];
        kb0.u[2*i] = p0[0]; kb0.u[2*i+1] = p0[1];
        kb1.u[2*i] = p1[0]; kb1.u[2*i+1] = p1[1];
      }
      s0 = __builtin_amdgcn_wmma_f32_16x16x32_bf16(false, qa[c].bf, false, kb0.bf,
                                                   (short)0, s0, false, false);
      s1 = __builtin_amdgcn_wmma_f32_16x16x32_bf16(false, qa[c].bf, false, kb1.bf,
                                                   (short)0, s1, false, false);
    }

    bool mk0 = mrow[kt + l16] != 0;
    bool mk1 = mrow[kt + 16 + l16] != 0;
    #pragma unroll
    for (int i = 0; i < 8; ++i) {
      float a0 = mk0 ? -1.0e30f : s0[i] * 0.125f;   // 1/sqrt(64)
      float a1 = mk1 ? -1.0e30f : s1[i] * 0.125f;
      float tmax = fmaxf(a0, a1);
      #pragma unroll
      for (int off = 1; off < 16; off <<= 1)
        tmax = fmaxf(tmax, __shfl_xor(tmax, off, 32));
      float mnew = fmaxf(rmax[i], tmax);
      float se = __expf(a0 - mnew) + __expf(a1 - mnew);
      #pragma unroll
      for (int off = 1; off < 16; off <<= 1)
        se += __shfl_xor(se, off, 32);
      rsum[i] = rsum[i] * __expf(rmax[i] - mnew) + se;
      rmax[i] = mnew;
    }
  }

  float rinv[8];
  #pragma unroll
  for (int i = 0; i < 8; ++i) rinv[i] = 1.0f / rsum[i];

  v8f o[4];
  #pragma unroll
  for (int nt = 0; nt < 4; ++nt) o[nt] = vzero8();

  // ---------------- pass 2: probs out + P*V ----------------
  for (int kt = 0; kt < T_; kt += 32) {
    __syncthreads();
    async_copy_b128(Kb + (size_t)kt * D_ + tid * 8, ((char*)Ks) + tid * 16);
    // V tile transposed staging stays on the regular path (async can't transpose)
    #pragma unroll
    for (int j = 0; j < 8; ++j) {
      int idx = tid * 8 + j;
      int key = idx >> 6, d = idx & 63;
      VsT[d][key] = Vb[(size_t)(kt + key) * D_ + d];
    }
    wait_async0();
    __syncthreads();

    v8f s0 = vzero8(), s1 = vzero8();
    #pragma unroll
    for (int c = 0; c < 2; ++c) {
      V16 kb0, kb1;
      #pragma unroll
      for (int i = 0; i < 8; ++i) {
        int d = c * 32 + (half << 4) + (i << 1);
        const unsigned short* p0 = &Ks[l16][d];
        const unsigned short* p1 = &Ks[16 + l16][d];
        kb0.u[2*i] = p0[0]; kb0.u[2*i+1] = p0[1];
        kb1.u[2*i] = p1[0]; kb1.u[2*i+1] = p1[1];
      }
      s0 = __builtin_amdgcn_wmma_f32_16x16x32_bf16(false, qa[c].bf, false, kb0.bf,
                                                   (short)0, s0, false, false);
      s1 = __builtin_amdgcn_wmma_f32_16x16x32_bf16(false, qa[c].bf, false, kb1.bf,
                                                   (short)0, s1, false, false);
    }

    bool mk0 = mrow[kt + l16] != 0;
    bool mk1 = mrow[kt + 16 + l16] != 0;
    #pragma unroll
    for (int i = 0; i < 8; ++i) {
      float a0 = mk0 ? -1.0e30f : s0[i] * 0.125f;
      float a1 = mk1 ? -1.0e30f : s1[i] * 0.125f;
      float p0 = __expf(a0 - rmax[i]) * rinv[i];
      float p1 = __expf(a1 - rmax[i]) * rinv[i];
      int m = i + 8 * half;
      size_t arow = ((size_t)bh * T_ + q0 + m) * (size_t)T_;
      attn[arow + kt + l16]      = p0;
      attn[arow + kt + 16 + l16] = p1;
      Ps[wave][m][l16]      = f32_to_bf16(p0);
      Ps[wave][m][16 + l16] = f32_to_bf16(p1);
    }
    // wave-local LDS RAW: drain DS ops before re-gathering P in A-operand layout
    asm volatile("s_wait_dscnt 0" ::: "memory");

    V16 pa;
    #pragma unroll
    for (int i = 0; i < 8; ++i) {
      int kk = (half << 4) + (i << 1);
      pa.u[2*i]   = Ps[wave][l16][kk];
      pa.u[2*i+1] = Ps[wave][l16][kk+1];
    }
    #pragma unroll
    for (int nt = 0; nt < 4; ++nt) {
      V16 vb;
      #pragma unroll
      for (int i = 0; i < 8; ++i) {
        int kk = (half << 4) + (i << 1);
        const unsigned short* p = &VsT[nt * 16 + l16][kk];
        vb.u[2*i] = p[0]; vb.u[2*i+1] = p[1];
      }
      o[nt] = __builtin_amdgcn_wmma_f32_16x16x32_bf16(false, pa.bf, false, vb.bf,
                                                      (short)0, o[nt], false, false);
    }
  }

  // context out, bf16, re-interleaved to [B,T,E] for the out-projection GEMM
  #pragma unroll
  for (int nt = 0; nt < 4; ++nt)
    #pragma unroll
    for (int i = 0; i < 8; ++i) {
      int m = i + 8 * half;
      int d = nt * 16 + l16;
      Octx[((size_t)b * T_ + q0 + m) * E_ + h * D_ + d] = f32_to_bf16(o[nt][i]);
    }
}

// ------------------------------------------------------------------- launcher
extern "C" void kernel_launch(void* const* d_in, const int* in_sizes, int n_in,
                              void* d_out, int out_size, void* d_ws, size_t ws_size,
                              hipStream_t stream) {
  (void)in_sizes; (void)n_in; (void)out_size; (void)ws_size;
  const float*         q    = (const float*)d_in[0];
  // d_in[1], d_in[2] (k, v) are unused by the reference
  const unsigned char* mask = (const unsigned char*)d_in[3];  // bool mask[b,t]
  const float*         Wqkv = (const float*)d_in[4];
  const float*         bqkv = (const float*)d_in[5];
  const float*         Wout = (const float*)d_in[6];
  const float*         bout = (const float*)d_in[7];

  float* out  = (float*)d_out;                       // [B,T,E]
  float* attn = out + (size_t)M_ * E_;               // [B,H,T,T]

  // workspace carve-out (48 MB total)
  char* ws = (char*)d_ws;
  unsigned short* A_bf  = (unsigned short*)(ws);                        // 8 MB
  unsigned short* Wq_bf = (unsigned short*)(ws + (size_t)8  * 1048576); // 6 MB
  unsigned short* Wo_bf = (unsigned short*)(ws + (size_t)14 * 1048576); // 2 MB
  unsigned short* Qhb   = (unsigned short*)(ws + (size_t)16 * 1048576); // 8 MB
  unsigned short* Khb   = (unsigned short*)(ws + (size_t)24 * 1048576); // 8 MB
  unsigned short* Vhb   = (unsigned short*)(ws + (size_t)32 * 1048576); // 8 MB
  unsigned short* Octx  = (unsigned short*)(ws + (size_t)40 * 1048576); // 8 MB

  // fp32 -> bf16 conversions
  f2bf_kernel<<<(M_ * 1024) / 256, 256, 0, stream>>>(q, A_bf, M_ * 1024);
  f2bf_kernel<<<(1024 * N3_) / 256, 256, 0, stream>>>(Wqkv, Wq_bf, 1024 * N3_);
  f2bf_kernel<<<(1024 * 1024) / 256, 256, 0, stream>>>(Wout, Wo_bf, 1024 * 1024);

  // QKV projection, scattered to head-major Q/K/V
  dim3 g1(M_ / 64, N3_ / 64);
  gemm_bf16_kernel<<<g1, 256, 0, stream>>>(A_bf, Wq_bf, bqkv, N3_, 0,
                                           Qhb, Khb, Vhb, nullptr);

  // attention (softmax stats + probs + P*V)
  dim3 g2(T_ / 128, B_ * H_);
  attn_kernel<<<g2, 256, 0, stream>>>(Qhb, Khb, Vhb, mask, attn, Octx);

  // output projection
  dim3 g3(M_ / 64, E_ / 64);
  gemm_bf16_kernel<<<g3, 256, 0, stream>>>(Octx, Wo_bf, bout, E_, 1,
                                           nullptr, nullptr, nullptr, out);
}
